// SABlock_51333449122132
// MI455X (gfx1250) — matrix operand order repaired and checked by reading, not compiled, not run on hardware
//
#include <hip/hip_runtime.h>
#include <hip/hip_bf16.h>

#define HIDDEN  384
#define NHEADS  6
#define HDIM    64
#define NTOK    4096
#define QKVDIM  (3*HIDDEN)   // 1152
#define SCALE   0.125f       // HDIM^-0.5

typedef __attribute__((ext_vector_type(16))) _Float16 v16h;
typedef __attribute__((ext_vector_type(8)))  _Float16 v8h;
typedef __attribute__((ext_vector_type(8)))  float    v8f;

union V16U { v16h v; v8h h2[2]; _Float16 h[16]; };

// ---- WMMA fragment helpers (wave32, V_WMMA_F32_16X16X32_F16) ----
// A (16x32, f16) lane layout: row = lane%16; halves[0..15] = K kb..kb+15,
// kb = 16*(lane/16).  Source row-major.
__device__ inline v16h load_a(const _Float16* __restrict__ base, int ld, int lane) {
  const int row = lane & 15;
  const int kb  = (lane >> 4) << 4;
  const _Float16* p = base + (size_t)row * ld + kb;
  V16U r;
  r.h2[0] = *(const v8h*)(p);
  r.h2[1] = *(const v8h*)(p + 8);
  return r.v;
}
// B (32x16, f16) lane layout: col = lane%16; halves[h] = K 16*(lane/16)+h
// (contiguous K per lane-half, cf. ISA sparse-B table). Source is the
// TRANSPOSED operand: rows = N (out col / query), contiguous K.
__device__ inline v16h load_b(const _Float16* __restrict__ base, int ld, int lane) {
  const int col = lane & 15;
  const int ko  = (lane >> 4) << 4;
  return *(const v16h*)(base + (size_t)col * ld + ko);
}
__device__ inline v8f wmma_f16(v16h a, v16h b, v8f c) {
  return __builtin_amdgcn_wmma_f32_16x16x32_f16(false, a, false, b, (short)0, c, false, false);
}

// ---- prep: fp32 -> f16 transposes ----
__global__ void k_xpose_x(const float* __restrict__ x, _Float16* __restrict__ xh) {
  int i = blockIdx.x * blockDim.x + threadIdx.x;       // over NTOK*HIDDEN
  if (i >= NTOK * HIDDEN) return;
  int n = i / HIDDEN, c = i % HIDDEN;
  xh[i] = (_Float16)x[(size_t)c * NTOK + n];           // x is [c][n]
}
__global__ void k_xpose_w(const float* __restrict__ w, _Float16* __restrict__ wt,
                          int K, int O) {               // w: [K][O] -> wt: [O][K]
  int i = blockIdx.x * blockDim.x + threadIdx.x;
  if (i >= K * O) return;
  int o = i / K, k = i % K;
  wt[i] = (_Float16)w[(size_t)k * O + o];
}
__global__ void k_xpose_v(const _Float16* __restrict__ qkv, _Float16* __restrict__ vt) {
  int i = blockIdx.x * blockDim.x + threadIdx.x;       // over HIDDEN*NTOK
  if (i >= HIDDEN * NTOK) return;
  int dc = i / NTOK, n = i % NTOK;                     // dc = head*64+d
  vt[i] = qkv[(size_t)n * QKVDIM + 2 * HIDDEN + dc];
}

// ---- GEMM: wave computes 32(M) x 64(N); B-frags shared across both A rows ----
__global__ __launch_bounds__(128)
void gemm_f16(const _Float16* __restrict__ A, const _Float16* __restrict__ Bt,
              _Float16* __restrict__ C, int K, int N) {
  const int lane = threadIdx.x & 31;
  const int wave = threadIdx.x >> 5;
  const int m0 = blockIdx.x * 128 + wave * 32;
  const int n0 = blockIdx.y * 64;
  v8f acc[2][4] = {};
  for (int k0 = 0; k0 < K; k0 += 32) {
    if (k0 + 32 < K) {
      // locality 3 -> WGP-scope prefetch: pulls into all cache levels
      __builtin_prefetch(A + (size_t)m0 * K + k0 + 32, 0, 3);
      __builtin_prefetch(Bt + (size_t)n0 * K + k0 + 32, 0, 3);
    }
    v16h a0 = load_a(A + (size_t)m0 * K + k0,        K, lane);
    v16h a1 = load_a(A + (size_t)(m0 + 16) * K + k0, K, lane);
    for (int j = 0; j < 4; ++j) {
      v16h b = load_b(Bt + (size_t)(n0 + j * 16) * K + k0, K, lane);
      acc[0][j] = wmma_f16(a0, b, acc[0][j]);
      acc[1][j] = wmma_f16(a1, b, acc[1][j]);
    }
  }
  const int rbase = (lane >> 4) << 3;
  for (int t = 0; t < 2; ++t)
    for (int j = 0; j < 4; ++j)
      for (int r = 0; r < 8; ++r)
        C[(size_t)(m0 + t * 16 + rbase + r) * N + n0 + j * 16 + (lane & 15)] =
            (_Float16)acc[t][j][r];
}

// ---- out-proj GEMM: f32 store + bias, [n][c] flat order ----
__global__ __launch_bounds__(128)
void gemm_out(const _Float16* __restrict__ A, const _Float16* __restrict__ Bt,
              const float* __restrict__ bias, float* __restrict__ C, int K, int N) {
  const int lane = threadIdx.x & 31;
  const int wave = threadIdx.x >> 5;
  const int m0 = blockIdx.x * 128 + wave * 32;
  const int n0 = blockIdx.y * 64;
  v8f acc[2][4] = {};
  for (int k0 = 0; k0 < K; k0 += 32) {
    if (k0 + 32 < K) {
      __builtin_prefetch(A + (size_t)m0 * K + k0 + 32, 0, 3);
      __builtin_prefetch(Bt + (size_t)n0 * K + k0 + 32, 0, 3);
    }
    v16h a0 = load_a(A + (size_t)m0 * K + k0,        K, lane);
    v16h a1 = load_a(A + (size_t)(m0 + 16) * K + k0, K, lane);
    for (int j = 0; j < 4; ++j) {
      v16h b = load_b(Bt + (size_t)(n0 + j * 16) * K + k0, K, lane);
      acc[0][j] = wmma_f16(a0, b, acc[0][j]);
      acc[1][j] = wmma_f16(a1, b, acc[1][j]);
    }
  }
  const int rbase = (lane >> 4) << 3;
  for (int t = 0; t < 2; ++t)
    for (int j = 0; j < 4; ++j) {
      int col = n0 + j * 16 + (lane & 15);
      float bv = bias[col];
      for (int r = 0; r < 8; ++r)
        C[(size_t)(m0 + t * 16 + rbase + r) * N + col] = acc[t][j][r] + bv;
    }
}

// ---- flash attention, transposed: S^T = K*Q^T, O^T = V^T*P^T ----
// One wave per (32-query block, head): two 16-query tiles share every K/V
// fragment load; online softmax is lane-local (per query column).
__global__ __launch_bounds__(32)
void attn_kernel(const _Float16* __restrict__ qkv, const _Float16* __restrict__ vt,
                 _Float16* __restrict__ oh) {
  const int lane = threadIdx.x;
  const int q0   = blockIdx.x * 32;
  const int head = blockIdx.y;

  // Q^T B-fragments (2 query tiles x dims {0-31, 32-63}), pre-scaled
  v16h qb[2][2];
  for (int qt = 0; qt < 2; ++qt) {
    const _Float16* qbase = qkv + (size_t)(q0 + qt * 16) * QKVDIM + head * HDIM;
    qb[qt][0] = load_b(qbase,      QKVDIM, lane);
    qb[qt][1] = load_b(qbase + 32, QKVDIM, lane);
    V16U u0, u1; u0.v = qb[qt][0]; u1.v = qb[qt][1];
    for (int i = 0; i < 16; ++i) {
      u0.h[i] = (_Float16)((float)u0.h[i] * SCALE);
      u1.h[i] = (_Float16)((float)u1.h[i] * SCALE);
    }
    qb[qt][0] = u0.v; qb[qt][1] = u1.v;
  }

  const _Float16* kbase = qkv + HIDDEN + head * HDIM;       // K block
  const _Float16* vtb   = vt + (size_t)head * HDIM * NTOK;  // V^T [d][n]

  float m[2] = {-1e30f, -1e30f}, l[2] = {0.f, 0.f};
  v8f o[2][4] = {};                                         // O^T: 64 d x 16 q

  for (int kc = 0; kc < NTOK; kc += 32) {
    if (kc + 32 < NTOK) {
      __builtin_prefetch(kbase + (size_t)(kc + 32) * QKVDIM, 0, 3);
      __builtin_prefetch(vtb + kc + 32, 0, 3);
    }
    // K A-fragments (16 keys x 32 dims), two key sub-tiles x two dim halves
    v16h ka0  = load_a(kbase + (size_t)kc * QKVDIM,             QKVDIM, lane);
    v16h ka0b = load_a(kbase + (size_t)kc * QKVDIM + 32,        QKVDIM, lane);
    v16h ka1  = load_a(kbase + (size_t)(kc + 16) * QKVDIM,      QKVDIM, lane);
    v16h ka1b = load_a(kbase + (size_t)(kc + 16) * QKVDIM + 32, QKVDIM, lane);
    // V^T A-fragments (4 d-blocks x 32 keys), shared by both query tiles
    v16h va[4];
    for (int f = 0; f < 4; ++f)
      va[f] = load_a(vtb + (size_t)(f * 16) * NTOK + kc, NTOK, lane);

    for (int qt = 0; qt < 2; ++qt) {
      v8f c0 = {}, c1 = {};
      c0 = wmma_f16(ka0,  qb[qt][0], c0);
      c0 = wmma_f16(ka0b, qb[qt][1], c0);
      c1 = wmma_f16(ka1,  qb[qt][0], c1);
      c1 = wmma_f16(ka1b, qb[qt][1], c1);

      // online softmax, per query column == per lane
      float mnew = m[qt];
      for (int r = 0; r < 8; ++r) mnew = fmaxf(mnew, fmaxf(c0[r], c1[r]));
      float alpha = __expf(m[qt] - mnew);
      float p0[8], p1[8], ps = 0.f;
      for (int r = 0; r < 8; ++r) {
        p0[r] = __expf(c0[r] - mnew);
        p1[r] = __expf(c1[r] - mnew);
        ps += p0[r] + p1[r];
      }
      l[qt] = l[qt] * alpha + ps;
      m[qt] = mnew;
      for (int f = 0; f < 4; ++f)
        for (int r = 0; r < 8; ++r) o[qt][f][r] *= alpha;

      // Build P^T (32 keys x 16 q) B-fragment: C-rows 8..15 live in the
      // opposite 16-lane half -> one xor-16 shuffle per register.
      V16U pb;
      for (int r = 0; r < 8; ++r) {
        float b0 = __shfl_xor(p0[r], 16, 32);
        float b1 = __shfl_xor(p1[r], 16, 32);
        if (lane < 16) { pb.h[r] = (_Float16)p0[r]; pb.h[8 + r] = (_Float16)b0; }
        else           { pb.h[r] = (_Float16)b1;    pb.h[8 + r] = (_Float16)p1[r]; }
      }

      // O^T += V^T * P^T  (4 d-blocks of 16)
      for (int f = 0; f < 4; ++f)
        o[qt][f] = wmma_f16(va[f], pb.v, o[qt][f]);
    }
  }

  const int rbase = (lane >> 4) << 3;
  for (int qt = 0; qt < 2; ++qt) {
    const float inv = 1.f / l[qt];
    const int q = q0 + qt * 16 + (lane & 15);
    for (int f = 0; f < 4; ++f)
      for (int r = 0; r < 8; ++r)
        oh[(size_t)q * HIDDEN + head * HDIM + f * 16 + rbase + r] =
            (_Float16)(o[qt][f][r] * inv);
  }
}

extern "C" void kernel_launch(void* const* d_in, const int* in_sizes, int n_in,
                              void* d_out, int out_size, void* d_ws, size_t ws_size,
                              hipStream_t stream) {
  const float* x    = (const float*)d_in[0];   // [1,384,16,16,16] == [c][n]
  const float* wqkv = (const float*)d_in[1];   // [384,1152]
  const float* wout = (const float*)d_in[2];   // [384,384]
  const float* bout = (const float*)d_in[3];   // [384]
  float* out = (float*)d_out;                  // flat [n][c]

  char* ws = (char*)d_ws;
  size_t off = 0;
  auto carve = [&](size_t bytes) -> char* {
    char* p = ws + off;
    off += (bytes + 255) & ~(size_t)255;
    return p;
  };
  _Float16* xh     = (_Float16*)carve((size_t)NTOK * HIDDEN * 2);
  _Float16* wqkvt  = (_Float16*)carve((size_t)QKVDIM * HIDDEN * 2);
  _Float16* woutt  = (_Float16*)carve((size_t)HIDDEN * HIDDEN * 2);
  _Float16* qkvb   = (_Float16*)carve((size_t)NTOK * QKVDIM * 2);
  _Float16* vtb    = (_Float16*)carve((size_t)HIDDEN * NTOK * 2);
  _Float16* ohb    = (_Float16*)carve((size_t)NTOK * HIDDEN * 2);

  const int tp = 256;
  k_xpose_x<<<(NTOK * HIDDEN + tp - 1) / tp, tp, 0, stream>>>(x, xh);
  k_xpose_w<<<(HIDDEN * QKVDIM + tp - 1) / tp, tp, 0, stream>>>(wqkv, wqkvt, HIDDEN, QKVDIM);
  k_xpose_w<<<(HIDDEN * HIDDEN + tp - 1) / tp, tp, 0, stream>>>(wout, woutt, HIDDEN, HIDDEN);

  gemm_f16<<<dim3(NTOK / 128, QKVDIM / 64), 128, 0, stream>>>(xh, wqkvt, qkvb, HIDDEN, QKVDIM);

  k_xpose_v<<<(HIDDEN * NTOK + tp - 1) / tp, tp, 0, stream>>>(qkvb, vtb);

  attn_kernel<<<dim3(NTOK / 32, NHEADS), 32, 0, stream>>>(qkvb, vtb, ohb);

  gemm_out<<<dim3(NTOK / 128, HIDDEN / 64), 128, 0, stream>>>(ohb, woutt, bout, out, HIDDEN, HIDDEN);
}
